// BasicCondConvBlock_34763465293958
// MI455X (gfx1250) — matrix operand order repaired and checked by reading, not compiled, not run on hardware
//
#include <hip/hip_runtime.h>

typedef __attribute__((ext_vector_type(16))) __bf16 v16bf;
typedef __attribute__((ext_vector_type(8)))  float  v8f;

__device__ __forceinline__ unsigned short f2bf(float f) {
    unsigned int u = __float_as_uint(f);
    unsigned int r = u + 0x7FFFu + ((u >> 16) & 1u);   // round-to-nearest-even
    return (unsigned short)(r >> 16);
}
__device__ __forceinline__ float bf2f(unsigned short h) {
    return __uint_as_float(((unsigned int)h) << 16);
}

// ---------------------------------------------------------------------------
// Routing: r[b] = sigmoid( (1/W) * sum_{ci,w} x[b,ci,w] * fcw[ci] + fcb )
// BF16IN=false : x layout [B][C][W] f32 ;  BF16IN=true : [B][W][C] bf16
// ---------------------------------------------------------------------------
template <bool BF16IN>
__global__ __launch_bounds__(256)
void route_kernel(const void* __restrict__ xin, const float* __restrict__ fcw,
                  const float* __restrict__ fcb, float* __restrict__ rout,
                  int C, int W) {
    const int b = blockIdx.x, tid = threadIdx.x;
    const int total = C * W;
    float p = 0.f;
    if (BF16IN) {
        const unsigned short* x = (const unsigned short*)xin + (size_t)b * total;
        for (int i = tid; i < total; i += 256) p += bf2f(x[i]) * fcw[i % C];
    } else {
        const float* x = (const float*)xin + (size_t)b * total;
        for (int i = tid; i < total; i += 256) p += x[i] * fcw[i / W];
    }
    #pragma unroll
    for (int off = 16; off > 0; off >>= 1) p += __shfl_down(p, off, 32);
    __shared__ float red[8];
    if ((tid & 31) == 0) red[tid >> 5] = p;
    __syncthreads();
    if (tid == 0) {
        float s = 0.f;
        #pragma unroll
        for (int i = 0; i < 8; ++i) s += red[i];
        float z = s / (float)W + fcb[0];
        rout[b] = 1.f / (1.f + __expf(-z));
    }
}

// ---------------------------------------------------------------------------
// CondConv1d(k=3,pad=1) + gate + BN(eval) + LeakyReLU(0.1) via bf16 WMMA.
//   FIRST:  input  f32 [B][CIN][W],  output bf16 [B][W][COUT]
//  !FIRST:  input bf16 [B][W][CIN] (async-copied to LDS), output f32
//           [B][COUT][W/2] with fused maxpool-2.
// Block tile: MT=64 Cout rows x NT=128 W cols, 8 waves (wave computes 16x64).
// LDS rows padded to CIN+8 elems: keeps every fragment 16B-aligned (b128
// ds loads + async b128 stores) and rotates banks by 4 dwords per row.
// ---------------------------------------------------------------------------
template <int CIN, int COUT, bool FIRST>
__global__ __launch_bounds__(256)
void condconv_wmma(const float* __restrict__ xf,
                   const unsigned short* __restrict__ xb,
                   const float* __restrict__ wgt,    // [COUT][CIN][3]
                   const float* __restrict__ bias,   // [COUT]
                   const float* __restrict__ g, const float* __restrict__ be,
                   const float* __restrict__ rm, const float* __restrict__ rv,
                   const float* __restrict__ rgate,  // [B]
                   unsigned short* __restrict__ ybf,
                   float* __restrict__ yout, int W) {
    constexpr int NT = 128, MT = 64, LROW = CIN + 8;
    extern __shared__ unsigned short smem[];
    unsigned short* xs  = smem;                       // [NT+2][LROW]  x^T tile
    unsigned short* wsh = smem + (NT + 2) * LROW;     // [3][MT][LROW] weights

    const int tid = threadIdx.x;
    const int w0  = blockIdx.x * NT;
    const int coB = blockIdx.y * MT;
    const int b   = blockIdx.z;

    if (tid == 0) __builtin_prefetch(wgt + (size_t)coB * CIN * 3, 0, 1);

    // stage weights -> bf16, layout [tap][m][ci] (K contiguous per tap)
    for (int i = tid; i < MT * CIN * 3; i += 256) {
        int m = i / (CIN * 3), rem = i - m * (CIN * 3);
        int ci = rem / 3, dk = rem - ci * 3;
        wsh[(dk * MT + m) * LROW + ci] = f2bf(wgt[(size_t)(coB + m) * CIN * 3 + rem]);
    }
    // stage X tile transposed with 1-col halo each side, zero padded
    if constexpr (FIRST) {
        for (int i = tid; i < CIN * (NT + 2); i += 256) {
            int ci = i / (NT + 2), wl = i - ci * (NT + 2);
            int gw = w0 - 1 + wl;
            float v = (gw >= 0 && gw < W) ? xf[((size_t)b * CIN + ci) * W + gw] : 0.f;
            xs[wl * LROW + ci] = f2bf(v);
        }
    } else {
        // async DMA: 16B (8 bf16 channels) per lane, global -> LDS, no VGPR data
        constexpr int SEG = CIN / 8;                 // 16B segments per row
        for (int t = tid; t < (NT + 2) * SEG; t += 256) {
            int wl = t / SEG, seg = t - wl * SEG;
            int gw = w0 - 1 + wl;
            unsigned short* dst = xs + wl * LROW + seg * 8;
            if (gw >= 0 && gw < W) {
                unsigned lds_off = (unsigned)(size_t)dst;
                unsigned long long gaddr =
                    (unsigned long long)(xb + ((size_t)b * W + gw) * CIN + seg * 8);
                asm volatile("global_load_async_to_lds_b128 %0, %1, off"
                             :: "v"(lds_off), "v"(gaddr) : "memory");
            } else {
                *(uint4*)dst = make_uint4(0u, 0u, 0u, 0u);
            }
        }
        asm volatile("s_wait_asynccnt 0x0" ::: "memory");
    }
    __syncthreads();

    const int lane = tid & 31;
    const int wv   = tid >> 5;
    const int cbL  = 16 * (wv & 3);   // wave's local Cout offset
    const int nb   = 64 * (wv >> 2);  // wave's local W offset
    const int nl   = lane & 15;
    const int half = lane >> 4;

    v8f acc[4] = {};
    union Frag { v16bf v; uint4 q[2]; };

    #pragma unroll
    for (int dk = 0; dk < 3; ++dk) {
        const unsigned short* wt = wsh + dk * MT * LROW;
        #pragma unroll
        for (int kc = 0; kc < CIN; kc += 32) {
            // A (16x32 bf16): VGPR0-3 = K 8h..8h+7, VGPR4-7 = K 16+8h..+7
            Frag A;
            const unsigned short* ar = wt + (cbL + nl) * LROW + kc + 8 * half;
            A.q[0] = *(const uint4*)(ar);
            A.q[1] = *(const uint4*)(ar + 16);
            #pragma unroll
            for (int nf = 0; nf < 4; ++nf) {
                // B (32x16 bf16): col N = lane&15, VGPR0-3 = K 8h..8h+7, VGPR4-7 = K 16+8h..+7
                Frag Bm;
                const unsigned short* br =
                    xs + (nb + 16 * nf + nl + dk) * LROW + kc + 8 * half;
                Bm.q[0] = *(const uint4*)(br);
                Bm.q[1] = *(const uint4*)(br + 16);
                acc[nf] = __builtin_amdgcn_wmma_f32_16x16x32_bf16(
                    false, A.v, false, Bm.v, (short)0, acc[nf], false, false);
            }
        }
    }

    // epilogue: y = lrelu( (r*(acc+bias) - rm) * g/sqrt(rv+eps) + be )
    const float rg = rgate[b];
    #pragma unroll
    for (int nf = 0; nf < 4; ++nf) {
        const int n = nb + 16 * nf + nl;   // C frag: N = lane&15, M = r + 8*half
        const int w = w0 + n;
        if constexpr (FIRST) {
            union { unsigned short s[8]; uint4 q; } pk;
            #pragma unroll
            for (int r = 0; r < 8; ++r) {
                int co = coB + cbL + 8 * half + r;
                float t = (acc[nf][r] + bias[co]) * rg;
                float sc = g[co] * rsqrtf(rv[co] + 1e-5f);
                float y = (t - rm[co]) * sc + be[co];
                pk.s[r] = f2bf(y >= 0.f ? y : 0.1f * y);
            }
            *(uint4*)(ybf + ((size_t)b * W + w) * COUT + coB + cbL + 8 * half) = pk.q;
        } else {
            #pragma unroll
            for (int r = 0; r < 8; ++r) {
                int co = coB + cbL + 8 * half + r;
                float t = (acc[nf][r] + bias[co]) * rg;
                float sc = g[co] * rsqrtf(rv[co] + 1e-5f);
                float y = (t - rm[co]) * sc + be[co];
                y = (y >= 0.f) ? y : 0.1f * y;
                float mx = fmaxf(y, __shfl_xor(y, 1, 32));   // pool pair (w, w^1)
                if ((n & 1) == 0)
                    yout[((size_t)b * COUT + co) * (W >> 1) + (w >> 1)] = mx;
            }
        }
    }
}

extern "C" void kernel_launch(void* const* d_in, const int* in_sizes, int n_in,
                              void* d_out, int out_size, void* d_ws, size_t ws_size,
                              hipStream_t stream) {
    const float* x    = (const float*)d_in[0];
    const float* w1   = (const float*)d_in[1];
    const float* b1   = (const float*)d_in[2];
    const float* fcw1 = (const float*)d_in[3];
    const float* fcb1 = (const float*)d_in[4];
    const float* g1   = (const float*)d_in[5];
    const float* be1  = (const float*)d_in[6];
    const float* rm1  = (const float*)d_in[7];
    const float* rv1  = (const float*)d_in[8];
    const float* w2   = (const float*)d_in[9];
    const float* b2   = (const float*)d_in[10];
    const float* fcw2 = (const float*)d_in[11];
    const float* fcb2 = (const float*)d_in[12];
    const float* g2   = (const float*)d_in[13];
    const float* be2  = (const float*)d_in[14];
    const float* rm2  = (const float*)d_in[15];
    const float* rv2  = (const float*)d_in[16];

    const int W = 2048, C0 = 64, C1 = 128, C2 = 256;
    const int B = in_sizes[0] / (C0 * W);

    float* r1 = (float*)d_ws;
    float* r2 = r1 + 64;
    unsigned short* y1 = (unsigned short*)((char*)d_ws + 1024);  // [B][W][C1] bf16

    route_kernel<false><<<B, 256, 0, stream>>>(x, fcw1, fcb1, r1, C0, W);

    size_t smem1 = ((size_t)130 * (C0 + 8) + 3 * 64 * (C0 + 8)) * sizeof(unsigned short);
    condconv_wmma<64, 128, true><<<dim3(W / 128, C1 / 64, B), 256, smem1, stream>>>(
        x, nullptr, w1, b1, g1, be1, rm1, rv1, r1, y1, nullptr, W);

    route_kernel<true><<<B, 256, 0, stream>>>(y1, fcw2, fcb2, r2, C1, W);

    size_t smem2 = ((size_t)130 * (C1 + 8) + 3 * 64 * (C1 + 8)) * sizeof(unsigned short);
    condconv_wmma<128, 256, false><<<dim3(W / 128, C2 / 64, B), 256, smem2, stream>>>(
        nullptr, y1, w2, b2, g2, be2, rm2, rv2, r2, nullptr, (float*)d_out, W);
}
